// MultiheadAttention_8297876816659
// MI455X (gfx1250) — matrix operand order repaired and checked by reading, not compiled, not run on hardware
//
#include <hip/hip_runtime.h>
#include <hip/hip_bf16.h>

// ---------------------------------------------------------------------------
// MI455X (gfx1250) multi-head attention forward.
// All matmuls go through v_wmma_f32_16x16x32_bf16 (wave32 WMMA, fp32 accum).
// Flash-attention inner kernel avoids materializing the 4*16*2048*2048 score
// tensor; 192MB L2 keeps bf16 intermediates resident between kernels.
// ---------------------------------------------------------------------------

typedef __bf16 bf16;
typedef __attribute__((ext_vector_type(16))) __bf16 v16bf;
typedef __attribute__((ext_vector_type(8)))  __bf16 v8bf;
typedef __attribute__((ext_vector_type(8)))  float  v8f;

static constexpr int Bb = 4, Ss = 2048, Dd = 1024, Hh = 16, HDd = 64, TDd = 1024;
static constexpr int Mrows = Bb * Ss;            // 8192
static constexpr float ATTN_SCALE = 0.125f;      // 1/sqrt(64)

__device__ __forceinline__ bf16 f2bf(float f) {
  unsigned u = __builtin_bit_cast(unsigned, f);
  u = (u + 0x7FFFu + ((u >> 16) & 1u)) >> 16;   // round-to-nearest-even
  unsigned short s = (unsigned short)u;
  return __builtin_bit_cast(bf16, s);
}

__device__ __forceinline__ v16bf cat8(v8bf lo, v8bf hi) {
  return __builtin_shufflevector(lo, hi, 0,1,2,3,4,5,6,7,8,9,10,11,12,13,14,15);
}

__device__ __forceinline__ v8f wmma_bf16(v16bf a, v16bf b, v8f c) {
  // D = A(16x32) * B(32x16) + C, fp32 accumulate
  return __builtin_amdgcn_wmma_f32_16x16x32_bf16(false, a, false, b, (short)0, c,
                                                 false, false);
}

// ---------------------------------------------------------------------------
// fp32 -> bf16 elementwise convert
// ---------------------------------------------------------------------------
__global__ __launch_bounds__(256) void cvt_f32_bf16(const float* __restrict__ in,
                                                    bf16* __restrict__ out, int n) {
  int i = blockIdx.x * 256 + threadIdx.x;
  if (i < n) out[i] = f2bf(in[i]);
}

// ---------------------------------------------------------------------------
// fp32 [rows][cols] -> bf16 [cols][rows] (transpose + convert), 32x32 LDS tile
// Puts weights in [N][K] order so WMMA B-fragments are contiguous 32B loads.
// ---------------------------------------------------------------------------
__global__ __launch_bounds__(256) void transpose_cvt(const float* __restrict__ in,
                                                     bf16* __restrict__ out,
                                                     int rows, int cols) {
  __shared__ float tile[32][33];
  const int c0 = blockIdx.x * 32, r0 = blockIdx.y * 32;
  const int tx = threadIdx.x & 31, ty = threadIdx.x >> 5;   // ty: 0..7
  #pragma unroll
  for (int i = ty; i < 32; i += 8) tile[i][tx] = in[(size_t)(r0 + i) * cols + c0 + tx];
  __syncthreads();
  #pragma unroll
  for (int i = ty; i < 32; i += 8)
    out[(size_t)(c0 + i) * rows + r0 + tx] = f2bf(tile[tx][i]);
}

// ---------------------------------------------------------------------------
// GEMM: C[M,N] = A[M,K](bf16,row-major) * Bt[N,K](bf16, pre-transposed) + bias
// Block: 256 threads = 8 waves (2 along M x 4 along N). Block tile 128x128,
// wave tile 64x32 = 4x2 WMMA accumulators. K-step 32.
// ---------------------------------------------------------------------------
template <bool OUT_F32>
__global__ __launch_bounds__(256) void gemm_wmma(const bf16* __restrict__ A,
                                                 const bf16* __restrict__ Bt,
                                                 const float* __restrict__ bias,
                                                 void* __restrict__ Cout,
                                                 int M, int N, int K) {
  __shared__ bf16 As[128 * 32];   // [m][k]
  __shared__ bf16 Bs[128 * 32];   // [n][k]
  const int tid  = threadIdx.x;
  const int wave = tid >> 5, lane = tid & 31;
  const int wm = wave & 1, wn = wave >> 1;        // wave grid 2x4
  const int lh = lane >> 4, l16 = lane & 15;      // half-wave, lane-in-16
  const int m0 = blockIdx.y * 128, n0 = blockIdx.x * 128;

  v8f acc[4][2];
  #pragma unroll
  for (int mt = 0; mt < 4; mt++)
    #pragma unroll
    for (int nt = 0; nt < 2; nt++) acc[mt][nt] = (v8f){};

  const int ldr = tid >> 1;            // 0..127
  const int ldc = (tid & 1) * 16;      // 0 or 16

  for (int k0 = 0; k0 < K; k0 += 32) {
    {
      const bf16* g = A + (size_t)(m0 + ldr) * K + k0 + ldc;
      *(v8bf*)(&As[ldr * 32 + ldc])     = *(const v8bf*)(g);
      *(v8bf*)(&As[ldr * 32 + ldc + 8]) = *(const v8bf*)(g + 8);
    }
    {
      const bf16* g = Bt + (size_t)(n0 + ldr) * K + k0 + ldc;
      *(v8bf*)(&Bs[ldr * 32 + ldc])     = *(const v8bf*)(g);
      *(v8bf*)(&Bs[ldr * 32 + ldc + 8]) = *(const v8bf*)(g + 8);
    }
    if (k0 + 32 < K) {   // warm L2/L0 for next K-step (global_prefetch_b8)
      __builtin_prefetch(A  + (size_t)(m0 + ldr) * K + k0 + 32, 0, 1);
      __builtin_prefetch(Bt + (size_t)(n0 + ldr) * K + k0 + 32, 0, 1);
    }
    __syncthreads();

    // A fragments: row = wm*64 + mt*16 + l16; K contiguous at lh*16
    v16bf af[4];
    const int klo = lh * 16;
    #pragma unroll
    for (int mt = 0; mt < 4; mt++) {
      const bf16* p = &As[(wm * 64 + mt * 16 + l16) * 32];
      af[mt] = cat8(*(const v8bf*)(p + klo), *(const v8bf*)(p + klo + 8));
    }
    // B fragments: n = wn*32 + nt*16 + l16; K contiguous at lh*16
    v16bf bfv[2];
    #pragma unroll
    for (int nt = 0; nt < 2; nt++)
      bfv[nt] = *(const v16bf*)(&Bs[(wn * 32 + nt * 16 + l16) * 32 + lh * 16]);

    #pragma unroll
    for (int mt = 0; mt < 4; mt++)
      #pragma unroll
      for (int nt = 0; nt < 2; nt++)
        acc[mt][nt] = wmma_bf16(af[mt], bfv[nt], acc[mt][nt]);
    __syncthreads();
  }

  // Epilogue: C/D layout -> lane holds col l16, rows r + 8*lh
  #pragma unroll
  for (int mt = 0; mt < 4; mt++)
    #pragma unroll
    for (int nt = 0; nt < 2; nt++) {
      const int n = n0 + wn * 32 + nt * 16 + l16;
      const float bv = bias[n];
      #pragma unroll
      for (int r = 0; r < 8; r++) {
        const int m = m0 + wm * 64 + mt * 16 + r + lh * 8;
        const float v = acc[mt][nt][r] + bv;
        if (OUT_F32) ((float*)Cout)[(size_t)m * N + n] = v;
        else         ((bf16*)Cout)[(size_t)m * N + n]  = f2bf(v);
      }
    }
}

// ---------------------------------------------------------------------------
// Flash attention: grid = (S/64, B*H), block = 128 (4 waves).
// Each block: 64 query rows of one head; streams 64-key tiles with online
// softmax. scores = Q*K^T via WMMA, P*V via WMMA (P staged bf16 in LDS).
// ---------------------------------------------------------------------------
__global__ __launch_bounds__(128) void flash_attn(const bf16* __restrict__ Q,
                                                  const bf16* __restrict__ Km,
                                                  const bf16* __restrict__ V,
                                                  bf16* __restrict__ Oattn) {
  __shared__ bf16 Ks[64 * 64];   // [key][d]
  __shared__ bf16 Vt[64 * 64];   // [d][key]  (transposed at load)
  __shared__ bf16 Ps[64 * 64];   // softmaxed probabilities, bf16

  const int tid = threadIdx.x, wave = tid >> 5, lane = tid & 31;
  const int lh = lane >> 4, l16 = lane & 15;
  const int b = blockIdx.y >> 4, h = blockIdx.y & 15;
  const int q0 = blockIdx.x * 64;
  const size_t base = ((size_t)b * Ss) * TDd + (size_t)h * HDd;  // row-major [B*S][TD]

  // Q fragments (rows wave*16 + l16), 2 K-steps over head_dim
  v16bf qf[2];
  {
    const bf16* qp = Q + base + (size_t)(q0 + wave * 16 + l16) * TDd;
    const int klo = lh * 16;
    #pragma unroll
    for (int ks = 0; ks < 2; ks++)
      qf[ks] = cat8(*(const v8bf*)(qp + ks * 32 + klo),
                    *(const v8bf*)(qp + ks * 32 + klo + 8));
  }

  float mrow[8], lrow[8];
  v8f oacc[4];
  #pragma unroll
  for (int r = 0; r < 8; r++) { mrow[r] = -1e30f; lrow[r] = 0.f; }
  #pragma unroll
  for (int nt = 0; nt < 4; nt++) oacc[nt] = (v8f){};

  const int ldr = tid >> 1;            // key row 0..63
  const int ldc = (tid & 1) * 32;      // d offset 0/32

  for (int kt = 0; kt < Ss / 64; ++kt) {
    // Stage K tile straight, V tile transposed
    {
      const bf16* g = Km + base + (size_t)(kt * 64 + ldr) * TDd + ldc;
      #pragma unroll
      for (int c8 = 0; c8 < 4; c8++)
        *(v8bf*)(&Ks[ldr * 64 + ldc + c8 * 8]) = *(const v8bf*)(g + c8 * 8);
    }
    {
      const bf16* g = V + base + (size_t)(kt * 64 + ldr) * TDd + ldc;
      #pragma unroll
      for (int c8 = 0; c8 < 4; c8++) {
        v8bf vv = *(const v8bf*)(g + c8 * 8);
        #pragma unroll
        for (int i = 0; i < 8; i++) Vt[(ldc + c8 * 8 + i) * 64 + ldr] = vv[i];
      }
    }
    __syncthreads();

    // scores: 16 query rows (this wave) x 64 keys = 4 WMMA tiles, K-dim = 64
    v8f sc[4];
    #pragma unroll
    for (int nt = 0; nt < 4; nt++) {
      v8f c = (v8f){};
      #pragma unroll
      for (int ks = 0; ks < 2; ks++) {
        v16bf kv = *(const v16bf*)(&Ks[(nt * 16 + l16) * 64 + ks * 32 + lh * 16]);
        c = wmma_bf16(qf[ks], kv, c);
      }
      sc[nt] = c;
    }

    // online softmax per row (rows r + 8*lh of this wave's 16-row tile)
    #pragma unroll
    for (int r = 0; r < 8; r++) {
      float mx = -1e30f;
      #pragma unroll
      for (int nt = 0; nt < 4; nt++) mx = fmaxf(mx, sc[nt][r] * ATTN_SCALE);
      #pragma unroll
      for (int off = 1; off < 16; off <<= 1) mx = fmaxf(mx, __shfl_xor(mx, off, 32));
      const float mnew  = fmaxf(mrow[r], mx);
      const float alpha = __expf(mrow[r] - mnew);
      float sum = 0.f;
      const int prow = wave * 16 + r + lh * 8;
      #pragma unroll
      for (int nt = 0; nt < 4; nt++) {
        const float p = __expf(sc[nt][r] * ATTN_SCALE - mnew);
        sum += p;
        Ps[prow * 64 + nt * 16 + l16] = f2bf(p);
      }
      #pragma unroll
      for (int off = 1; off < 16; off <<= 1) sum += __shfl_xor(sum, off, 32);
      lrow[r] = lrow[r] * alpha + sum;
      mrow[r] = mnew;
      #pragma unroll
      for (int nt = 0; nt < 4; nt++) oacc[nt][r] *= alpha;
    }
    __syncthreads();

    // O += P * V : A = P rows (this wave), B = Vt (K=keys contiguous)
    {
      const bf16* pp = &Ps[(wave * 16 + l16) * 64];
      const int klo = lh * 16;
      v16bf pf[2];
      #pragma unroll
      for (int ks = 0; ks < 2; ks++)
        pf[ks] = cat8(*(const v8bf*)(pp + ks * 32 + klo),
                      *(const v8bf*)(pp + ks * 32 + klo + 8));
      #pragma unroll
      for (int nt = 0; nt < 4; nt++)
        #pragma unroll
        for (int ks = 0; ks < 2; ks++) {
          v16bf vf = *(const v16bf*)(&Vt[(nt * 16 + l16) * 64 + ks * 32 + lh * 16]);
          oacc[nt] = wmma_bf16(pf[ks], vf, oacc[nt]);
        }
    }
    __syncthreads();   // protect Ks/Vt/Ps before next tile overwrites
  }

  // epilogue: O / l  -> attn[B*S][TD] (bf16)
  #pragma unroll
  for (int nt = 0; nt < 4; nt++)
    #pragma unroll
    for (int r = 0; r < 8; r++) {
      const int q = q0 + wave * 16 + r + lh * 8;
      Oattn[base + (size_t)q * TDd + nt * 16 + l16] = f2bf(oacc[nt][r] / lrow[r]);
    }
}

// ---------------------------------------------------------------------------
extern "C" void kernel_launch(void* const* d_in, const int* in_sizes, int n_in,
                              void* d_out, int out_size, void* d_ws, size_t ws_size,
                              hipStream_t stream) {
  (void)in_sizes; (void)n_in; (void)out_size; (void)ws_size;
  const float* x  = (const float*)d_in[0];
  const float* Wq = (const float*)d_in[1];
  const float* bq = (const float*)d_in[2];
  const float* Wk = (const float*)d_in[3];
  const float* bk = (const float*)d_in[4];
  const float* Wv = (const float*)d_in[5];
  const float* bv = (const float*)d_in[6];
  const float* Wo = (const float*)d_in[7];
  const float* bo = (const float*)d_in[8];
  float* out = (float*)d_out;

  // Workspace carve-up (~88 MB)
  size_t off = 0;
  auto take = [&](size_t bytes) {
    char* p = (char*)d_ws + off;
    off = (off + bytes + 255) & ~(size_t)255;
    return p;
  };
  bf16* xb   = (bf16*)take((size_t)Mrows * Dd * 2);
  bf16* Wqt  = (bf16*)take((size_t)Dd * TDd * 2);
  bf16* Wkt  = (bf16*)take((size_t)Dd * TDd * 2);
  bf16* Wvt  = (bf16*)take((size_t)Dd * TDd * 2);
  bf16* Wot  = (bf16*)take((size_t)TDd * Dd * 2);
  bf16* Qb   = (bf16*)take((size_t)Mrows * TDd * 2);
  bf16* Kb   = (bf16*)take((size_t)Mrows * TDd * 2);
  bf16* Vb   = (bf16*)take((size_t)Mrows * TDd * 2);
  bf16* Attn = (bf16*)take((size_t)Mrows * TDd * 2);

  // 1) x -> bf16
  {
    int n = Mrows * Dd;
    cvt_f32_bf16<<<dim3((n + 255) / 256), dim3(256), 0, stream>>>(x, xb, n);
  }
  // 2) weights -> bf16, transposed to [N][K]
  {
    dim3 g(TDd / 32, Dd / 32), blk(256);
    transpose_cvt<<<g, blk, 0, stream>>>(Wq, Wqt, Dd, TDd);
    transpose_cvt<<<g, blk, 0, stream>>>(Wk, Wkt, Dd, TDd);
    transpose_cvt<<<g, blk, 0, stream>>>(Wv, Wvt, Dd, TDd);
    transpose_cvt<<<dim3(Dd / 32, TDd / 32), blk, 0, stream>>>(Wo, Wot, TDd, Dd);
  }
  // 3) Q/K/V projections (bias fused, bf16 out)
  {
    dim3 g(TDd / 128, Mrows / 128), blk(256);
    gemm_wmma<false><<<g, blk, 0, stream>>>(xb, Wqt, bq, Qb, Mrows, TDd, Dd);
    gemm_wmma<false><<<g, blk, 0, stream>>>(xb, Wkt, bk, Kb, Mrows, TDd, Dd);
    gemm_wmma<false><<<g, blk, 0, stream>>>(xb, Wvt, bv, Vb, Mrows, TDd, Dd);
  }
  // 4) flash attention per (query-tile, b*h)
  flash_attn<<<dim3(Ss / 64, Bb * Hh), dim3(128), 0, stream>>>(Qb, Kb, Vb, Attn);
  // 5) output projection (fp32 out = d_out)
  gemm_wmma<true><<<dim3(Dd / 128, Mrows / 128), dim3(256), 0, stream>>>(
      Attn, Wot, bo, out, Mrows, Dd, TDd);
}
